// ScaledDotProductAttention_78683800863253
// MI455X (gfx1250) — matrix operand order repaired and checked by reading, not compiled, not run on hardware
//
#include <hip/hip_runtime.h>

typedef __attribute__((ext_vector_type(16))) _Float16 v16h;
typedef __attribute__((ext_vector_type(8)))  float    v8f;
typedef __attribute__((ext_vector_type(4)))  float    v4f;

#define SEQ   2048
#define DH    64
#define SROW  2064            // padded score-row stride in floats (2048 + 16) -> no LDS bank conflicts
#define NKT   (SEQ / 16)      // 128 k-tiles of 16
#define BH    (2 * 16)        // batch*heads

__global__ __launch_bounds__(128) void sdpa_wmma_kernel(
    const float* __restrict__ Q,
    const float* __restrict__ K,
    const float* __restrict__ V,
    float* __restrict__ Out,        // [BH, SEQ, DH]
    float* __restrict__ Attn)       // [BH, SEQ, SEQ]
{
    extern __shared__ char smem_raw[];
    float*    sc  = (float*)smem_raw;                 // [16][SROW] f32 scores/probs
    _Float16* qh  = (_Float16*)(sc + 16 * SROW);      // [16][DH]   f16 staged Q tile
    float*    red = (float*)(qh + 16 * DH);           // rowmax[16] rowsum[16] part[128]
    float* rowmax = red;
    float* rowsum = red + 16;
    float* part   = red + 32;

    const int tid = threadIdx.x;
    const int l   = tid & 31;     // lane in wave32
    const int w   = tid >> 5;     // wave id 0..3
    const int m   = l & 15;       // M (A) / N (B,C,D) index
    const int g   = l >> 4;       // lane half-group

    const int qt = blockIdx.x & (NKT - 1);   // q tile 0..127
    const int bh = blockIdx.x >> 7;          // batch*head slice

    const size_t slice = (size_t)bh * SEQ * DH;
    const float* Qs = Q + slice + (size_t)qt * 16 * DH;
    const float* Ks = K + slice;
    const float* Vs = V + slice;
    float*       Os = Out + slice + (size_t)qt * 16 * DH;
    float*       As = Attn + (size_t)bh * SEQ * SEQ + (size_t)qt * 16 * SEQ;

    // ---- stage Q tile (16 x 64) as f16 in LDS ----
    for (int i = tid; i < 16 * DH; i += 128)
        qh[i] = (_Float16)Qs[i];
    __syncthreads();

    // ---- build the two Q A-fragments (K blocks [0,32) and [32,64)) ----
    // A k order used here (contiguous, same as B): halves 0..15 -> K = g*16+h
    v16h aq0, aq1;
    {
        const _Float16* qr = qh + m * DH;
#pragma unroll
        for (int h = 0; h < 8; ++h) {
            aq0[h]     = qr[g * 16 + h];
            aq0[8 + h] = qr[g * 16 + 8 + h];
            aq1[h]     = qr[32 + g * 16 + h];
            aq1[8 + h] = qr[32 + g * 16 + 8 + h];
        }
    }

    // ---- Phase 1: scores = scale * Q @ K^T  (per-wave strided over k-tiles) ----
    const float scale = 0.125f;   // 1/sqrt(64)
    for (int kt = w; kt < NKT; kt += 4) {
        // B fragment: B[kd][n] = Key[kt*16 + n][kd]; halves h -> K = kb + g*16 + h
        const float* kr = Ks + (size_t)(kt * 16 + m) * DH;   // n == m expression-wise
        const float* c0 = kr + g * 16;         // K block [0,32)
        const float* c1 = kr + 32 + g * 16;    // K block [32,64)
        v16h b0, b1;
#pragma unroll
        for (int h = 0; h < 16; ++h) {
            b0[h] = (_Float16)c0[h];
            b1[h] = (_Float16)c1[h];
        }
        v8f acc = {};
        acc = __builtin_amdgcn_wmma_f32_16x16x32_f16(false, aq0, false, b0,
                                                     (short)0, acc, false, false);
        acc = __builtin_amdgcn_wmma_f32_16x16x32_f16(false, aq1, false, b1,
                                                     (short)0, acc, false, false);
        // C/D layout: vgpr r -> M = r + 8*g, N = m
#pragma unroll
        for (int r = 0; r < 8; ++r)
            sc[(r + 8 * g) * SROW + kt * 16 + m] = acc[r] * scale;
    }
    __syncthreads();

    // ---- Phase 2: softmax over each of the 16 rows (f32, in LDS) ----
    {
        const int r   = tid >> 3;  // row 0..15
        const int sub = tid & 7;   // 8 partials per row

        float mx = -3.4e38f;
        for (int c = sub; c < SEQ; c += 8)
            mx = fmaxf(mx, sc[r * SROW + c]);
        part[r * 8 + sub] = mx;
        __syncthreads();
        if (sub == 0) {
            float v = part[r * 8];
#pragma unroll
            for (int j = 1; j < 8; ++j) v = fmaxf(v, part[r * 8 + j]);
            rowmax[r] = v;
        }
        __syncthreads();

        const float rm = rowmax[r];
        float sum = 0.0f;
        for (int c = sub; c < SEQ; c += 8) {
            float e = __expf(sc[r * SROW + c] - rm);
            sc[r * SROW + c] = e;
            sum += e;
        }
        part[r * 8 + sub] = sum;
        __syncthreads();
        if (sub == 0) {
            float v = 0.0f;
#pragma unroll
            for (int j = 0; j < 8; ++j) v += part[r * 8 + j];
            rowsum[r] = 1.0f / v;   // store reciprocal
        }
        __syncthreads();
    }

    // ---- Phase 3: normalize, stream attn weights out (dominant HBM traffic) ----
    // float4 + non-temporal stores: b128 stores, don't trash L2 (K/V reuse lives there)
    for (int i = tid; i < 16 * (SEQ / 4); i += 128) {
        const int rr = i >> 9;             // SEQ/4 = 512 float4 per row
        const int cc = (i & 511) * 4;
        v4f p = *(const v4f*)(sc + rr * SROW + cc);    // ds_load_b128
        p *= rowsum[rr];
        *(v4f*)(sc + rr * SROW + cc) = p;              // keep normalized P for PV
        __builtin_nontemporal_store(p, (v4f*)(As + (size_t)rr * SEQ + cc));
    }
    __syncthreads();

    // ---- Phase 4: output = P @ V ; each wave owns one 16-wide d-column tile ----
    const int d0 = w * 16;
    v8f oacc = {};
    for (int k0 = 0; k0 < SEQ; k0 += 32) {
        // A fragment from normalized P (f32 in LDS -> f16)
        v16h pa;
        const float* pr = sc + m * SROW + k0;
#pragma unroll
        for (int h = 0; h < 8; ++h) {
            pa[h]     = (_Float16)pr[g * 16 + h];
            pa[8 + h] = (_Float16)pr[g * 16 + 8 + h];
        }
        // B fragment from V: B[kk][n] = V[k0 + kk][d0 + n]; halves h -> kk = g*16 + h
        v16h vb;
        const float* vc = Vs + (size_t)(k0 + g * 16) * DH + d0 + m;
#pragma unroll
        for (int h = 0; h < 16; ++h)
            vb[h] = (_Float16)vc[(size_t)h * DH];
        oacc = __builtin_amdgcn_wmma_f32_16x16x32_f16(false, pa, false, vb,
                                                      (short)0, oacc, false, false);
    }
#pragma unroll
    for (int r = 0; r < 8; ++r)
        Os[(size_t)(r + 8 * g) * DH + d0 + m] = oacc[r];
}

extern "C" void kernel_launch(void* const* d_in, const int* in_sizes, int n_in,
                              void* d_out, int out_size, void* d_ws, size_t ws_size,
                              hipStream_t stream) {
    (void)in_sizes; (void)n_in; (void)out_size; (void)d_ws; (void)ws_size;
    const float* Q = (const float*)d_in[0];
    const float* K = (const float*)d_in[1];
    const float* V = (const float*)d_in[2];
    float* Out  = (float*)d_out;
    float* Attn = Out + (size_t)BH * SEQ * DH;   // outputs concatenated: [out, attn_weights]

    const size_t smem = (size_t)16 * SROW * sizeof(float)   // scores
                      + (size_t)16 * DH * 2                  // f16 Q tile
                      + (size_t)(32 + 128) * sizeof(float);  // reductions
    dim3 grid(BH * NKT);   // 2*16 * 128 = 4096 blocks
    sdpa_wmma_kernel<<<grid, 128, smem, stream>>>(Q, K, V, Out, Attn);
}
